// StructureModule_60936995996126
// MI455X (gfx1250) — matrix operand order repaired and checked by reading, not compiled, not run on hardware
//
#include <hip/hip_runtime.h>
#include <hip/hip_bf16.h>
#include <math.h>

typedef __attribute__((ext_vector_type(16))) __bf16 bf16x16;
typedef __attribute__((ext_vector_type(8)))  float  f32x8;

#define NRES 512
#define CS   384
#define CZ   128
#define NH   12
#define TSF  10.0f
#define EPSV 1e-8f
#define INFV 100000.0f

static inline int cdiv_h(long a, long b){ return (int)((a + b - 1) / b); }

// Fragment loads, contiguous k = 16*half + j in both A and B (16-bit A 16x32, B 32x16).
static __device__ __forceinline__ void loadA_frag(bf16x16& f, const __bf16* Ar, int k0, int half){
  const __bf16* p0 = Ar + k0 + half*16;
  __builtin_prefetch(p0 + 96, 0, 1);
  #pragma unroll
  for (int j = 0; j < 16; ++j) f[j] = p0[j];
}
static __device__ __forceinline__ void loadB_frag(bf16x16& f, const __bf16* Bc, int k0, int half){
  const __bf16* q0 = Bc + k0 + half*16;
  __builtin_prefetch(q0 + 96, 0, 1);
  #pragma unroll
  for (int j = 0; j < 16; ++j) f[j] = q0[j];
}

// ---------------------------------------------------------------------------
// WMMA GEMM: C[M,N] = A[M,K](bf16,row-major) * Bt[Npad,K](bf16)^T
// One wave computes a 16x32 output tile (two accumulators sharing the A
// fragment). Software-pipelined: fragments for step k+32 are loaded before
// the WMMAs for step k issue, so loads overlap the matrix pipe.  Loop is
// versioned on has2 so the steady state has no scalar control flow.  Caller
// guarantees A rows [0, gridDim.y*16) are loadable (padded buffers).
// ---------------------------------------------------------------------------
__global__ void gemm_bf16_wmma(const __bf16* __restrict__ A, int lda,
                               const __bf16* __restrict__ Bt,
                               float* __restrict__ C, int ldc,
                               int M, int Nreal, int K, int Npad,
                               long sA, long sB, long sC,
                               const float* __restrict__ bias,
                               const float* __restrict__ res, int ldres,
                               int relu)
{
  const int lane = threadIdx.x & 31;
  const int nb = blockIdx.x * 32;
  const int mb = blockIdx.y * 16;
  const int bz = blockIdx.z;
  A  += (long)bz * sA;
  Bt += (long)bz * sB;
  C  += (long)bz * sC;
  const float* rr = res ? (res + (long)bz * sC) : (const float*)0;

  const int half = lane >> 4;     // 0: lanes 0-15, 1: lanes 16-31
  const int l16  = lane & 15;
  const __bf16* Ar  = A + (long)(mb + l16) * lda;
  const __bf16* Bc0 = Bt + (long)(nb + l16) * K;        // Bt row stride == K
  const bool has2 = (nb + 16) < Npad;
  const __bf16* Bc1 = Bt + (long)(nb + 16 + l16) * K;

  f32x8 acc0 = {0.f,0.f,0.f,0.f,0.f,0.f,0.f,0.f};
  f32x8 acc1 = {0.f,0.f,0.f,0.f,0.f,0.f,0.f,0.f};
  bf16x16 af, bf0, bf1;

  if (has2) {
    loadA_frag(af, Ar, 0, half);
    loadB_frag(bf0, Bc0, 0, half);
    loadB_frag(bf1, Bc1, 0, half);
    int k0 = 0;
    for (; k0 + 32 < K; k0 += 32) {
      bf16x16 afn, bf0n, bf1n;
      loadA_frag(afn, Ar, k0 + 32, half);
      loadB_frag(bf0n, Bc0, k0 + 32, half);
      loadB_frag(bf1n, Bc1, k0 + 32, half);
      acc0 = __builtin_amdgcn_wmma_f32_16x16x32_bf16(false, af, false, bf0,
                                                     (short)0, acc0, false, false);
      acc1 = __builtin_amdgcn_wmma_f32_16x16x32_bf16(false, af, false, bf1,
                                                     (short)0, acc1, false, false);
      af = afn; bf0 = bf0n; bf1 = bf1n;
    }
    acc0 = __builtin_amdgcn_wmma_f32_16x16x32_bf16(false, af, false, bf0,
                                                   (short)0, acc0, false, false);
    acc1 = __builtin_amdgcn_wmma_f32_16x16x32_bf16(false, af, false, bf1,
                                                   (short)0, acc1, false, false);
  } else {
    loadA_frag(af, Ar, 0, half);
    loadB_frag(bf0, Bc0, 0, half);
    int k0 = 0;
    for (; k0 + 32 < K; k0 += 32) {
      bf16x16 afn, bf0n;
      loadA_frag(afn, Ar, k0 + 32, half);
      loadB_frag(bf0n, Bc0, k0 + 32, half);
      acc0 = __builtin_amdgcn_wmma_f32_16x16x32_bf16(false, af, false, bf0,
                                                     (short)0, acc0, false, false);
      af = afn; bf0 = bf0n;
    }
    acc0 = __builtin_amdgcn_wmma_f32_16x16x32_bf16(false, af, false, bf0,
                                                   (short)0, acc0, false, false);
  }

  #pragma unroll
  for (int tile = 0; tile < 2; ++tile) {
    if (tile == 1 && !has2) break;
    const int n = nb + tile*16 + l16;
    if (n < Nreal) {
      const float bv = bias ? bias[n] : 0.f;
      #pragma unroll
      for (int r = 0; r < 8; ++r) {
        const int m = mb + half*8 + r;
        if (m < M) {
          float v = (tile ? acc1[r] : acc0[r]) + bv;
          if (rr) v += rr[(long)m * ldres + n];
          if (relu) v = v > 0.f ? v : 0.f;
          C[(long)m * ldc + n] = v;
        }
      }
    }
  }
}

// ---------------------------------------------------------------------------
// LayerNorm over rows; optional f32 and bf16 outputs (in-place safe).
// ---------------------------------------------------------------------------
__global__ void ln_kernel(const float* __restrict__ x, const float* __restrict__ g,
                          const float* __restrict__ b, float* __restrict__ of32,
                          __bf16* __restrict__ obf, int cols)
{
  const long row = blockIdx.x;
  const float* xr = x + row * cols;
  __shared__ float red[128];
  float s = 0.f;
  for (int c = threadIdx.x; c < cols; c += 128) s += xr[c];
  red[threadIdx.x] = s; __syncthreads();
  for (int o = 64; o > 0; o >>= 1){ if (threadIdx.x < o) red[threadIdx.x] += red[threadIdx.x+o]; __syncthreads(); }
  const float mu = red[0] / cols; __syncthreads();
  float v = 0.f;
  for (int c = threadIdx.x; c < cols; c += 128){ float d = xr[c]-mu; v += d*d; }
  red[threadIdx.x] = v; __syncthreads();
  for (int o = 64; o > 0; o >>= 1){ if (threadIdx.x < o) red[threadIdx.x] += red[threadIdx.x+o]; __syncthreads(); }
  const float inv = rsqrtf(red[0]/cols + 1e-5f);
  for (int c = threadIdx.x; c < cols; c += 128){
    float y = (xr[c]-mu)*inv*g[c] + b[c];
    if (of32) of32[row*cols + c] = y;
    if (obf)  obf [row*cols + c] = (__bf16)y;
  }
}

__global__ void conv_bf16_kernel(const float* __restrict__ src, __bf16* __restrict__ dst,
                                 int n, int relu)
{
  int i = blockIdx.x*256 + threadIdx.x;
  if (i < n){ float v = src[i]; if (relu && v < 0.f) v = 0.f; dst[i] = (__bf16)v; }
}

__global__ void zero_bf16_kernel(__bf16* __restrict__ dst, int n)
{
  int i = blockIdx.x*256 + threadIdx.x;
  if (i < n) dst[i] = (__bf16)0.0f;
}

// (K,N) f32 -> (Npad,K) bf16 transpose, zero pad n>=N
__global__ void wtrans_kernel(const float* __restrict__ src, __bf16* __restrict__ dst,
                              int K, int N, int Npad)
{
  long idx = (long)blockIdx.x*256 + threadIdx.x;
  long tot = (long)Npad*K;
  if (idx < tot){
    int k = (int)(idx % K); int n = (int)(idx / K);
    dst[idx] = (n < N) ? (__bf16)src[(long)k*N + n] : (__bf16)0.0f;
  }
}

// z_bf[i][j][c] -> z_T[i][c][j]
__global__ void ztrans_kernel(const __bf16* __restrict__ zb, __bf16* __restrict__ zt)
{
  long idx = (long)blockIdx.x*256 + threadIdx.x;
  if (idx < (long)NRES*CZ*NRES){
    int j = (int)(idx & (NRES-1)); long r = idx >> 9;
    int c = (int)(r & (CZ-1)); int i = (int)(r >> 7);
    zt[idx] = zb[((long)i*NRES + j)*CZ + c];
  }
}

__global__ void copy_kernel(const float* __restrict__ s, float* __restrict__ d, int n)
{
  int i = blockIdx.x*256 + threadIdx.x;
  if (i < n) d[i] = s[i];
}

__global__ void init_qt(float* __restrict__ quat, float* __restrict__ tvec)
{
  int i = blockIdx.x*256 + threadIdx.x;
  if (i < NRES){
    quat[i*4+0]=1.f; quat[i*4+1]=0.f; quat[i*4+2]=0.f; quat[i*4+3]=0.f;
    tvec[i*3+0]=0.f; tvec[i*3+1]=0.f; tvec[i*3+2]=0.f;
  }
}

// ---------------------------------------------------------------------------
// Per-residue attention input prep: rotation from quat, global-frame points,
// q2/k2 row/col offsets, hw-scaled extended Q/K (K-dim padded to 32), and the
// transposed [v | v_pts] B-operand for the value GEMM.
// ---------------------------------------------------------------------------
__global__ void build_attn(const float* __restrict__ qry,  // 512x192
                           const float* __restrict__ kvf,  // 512x384
                           const float* __restrict__ qpf,  // 512x144
                           const float* __restrict__ kvpf, // 512x432
                           const float* __restrict__ quat,
                           const float* __restrict__ tvec,
                           const float* __restrict__ hwp,
                           float* __restrict__ Rbuf,
                           __bf16* __restrict__ Qext, __bf16* __restrict__ Kext,
                           float* __restrict__ q2, float* __restrict__ k2,
                           __bf16* __restrict__ VVt)
{
  const int i = blockIdx.x;
  const int tid = threadIdx.x; // 64
  __shared__ float sR[9], st[3], sqp[48*3], skp[144*3];
  if (tid == 0){
    float w=quat[i*4], x=quat[i*4+1], y=quat[i*4+2], z=quat[i*4+3];
    sR[0]=1-2*(y*y+z*z); sR[1]=2*(x*y-w*z); sR[2]=2*(x*z+w*y);
    sR[3]=2*(x*y+w*z);   sR[4]=1-2*(x*x+z*z); sR[5]=2*(y*z-w*x);
    sR[6]=2*(x*z-w*y);   sR[7]=2*(y*z+w*x);   sR[8]=1-2*(x*x+y*y);
    #pragma unroll
    for (int r=0;r<9;++r) Rbuf[i*9+r]=sR[r];
    st[0]=tvec[i*3]; st[1]=tvec[i*3+1]; st[2]=tvec[i*3+2];
  }
  __syncthreads();
  for (int p = tid; p < 48; p += 64){
    float lx=qpf[i*144+p], ly=qpf[i*144+48+p], lz=qpf[i*144+96+p];
    sqp[p*3+0]=sR[0]*lx+sR[1]*ly+sR[2]*lz+st[0];
    sqp[p*3+1]=sR[3]*lx+sR[4]*ly+sR[5]*lz+st[1];
    sqp[p*3+2]=sR[6]*lx+sR[7]*ly+sR[8]*lz+st[2];
  }
  for (int p = tid; p < 144; p += 64){
    float lx=kvpf[i*432+p], ly=kvpf[i*432+144+p], lz=kvpf[i*432+288+p];
    skp[p*3+0]=sR[0]*lx+sR[1]*ly+sR[2]*lz+st[0];
    skp[p*3+1]=sR[3]*lx+sR[4]*ly+sR[5]*lz+st[1];
    skp[p*3+2]=sR[6]*lx+sR[7]*ly+sR[8]*lz+st[2];
  }
  __syncthreads();
  if (tid < NH){
    int h = tid; float a=0.f, b=0.f;
    for (int p=0;p<4;++p)
      for (int d=0;d<3;++d){
        float qv = sqp[(h*4+p)*3+d];  a += qv*qv;
        float kv = skp[(h*12+p)*3+d]; b += kv*kv;
      }
    q2[i*NH+h]=a; k2[i*NH+h]=b;
  }
  for (int idx = tid; idx < NH*32; idx += 64){
    int h = idx>>5, c = idx&31;
    float hw = log1pf(expf(hwp[h])) * 0.13608276348795434f; // softplus * sqrt(1/54)
    float qv=0.f, kv=0.f;
    if (c < 16){ qv = qry[i*192 + h*16 + c]*0.14433756729740643f; kv = kvf[i*384 + h*32 + c]; }
    else if (c < 28){ int pc=c-16, p=pc/3, d=pc%3; qv = sqp[(h*4+p)*3+d]*hw; kv = skp[(h*12+p)*3+d]; }
    Qext[((long)h*NRES+i)*32+c]=(__bf16)qv;
    Kext[((long)h*NRES+i)*32+c]=(__bf16)kv;
  }
  for (int idx = tid; idx < NH*48; idx += 64){
    int h = idx/48, n = idx%48;
    float v=0.f;
    if (n < 16) v = kvf[i*384 + h*32 + 16 + n];
    else if (n < 40){ int pc=n-16, p=pc/3, d=pc%3; v = skp[(h*12+4+p)*3+d]; }
    VVt[((long)h*48+n)*NRES + i] = (__bf16)v;
  }
}

// ---------------------------------------------------------------------------
// Softmax row (h,i): add pair bias, -0.5*hw*(q2+k2), mask; write bf16 probs.
// ---------------------------------------------------------------------------
__global__ void attn_softmax(const float* __restrict__ logits,
                             const float* __restrict__ biasb,   // (i*512+j)*12+h
                             const float* __restrict__ q2,
                             const float* __restrict__ k2,
                             const float* __restrict__ hwp,
                             const float* __restrict__ mask,
                             __bf16* __restrict__ abf)
{
  const int i = blockIdx.x, h = blockIdx.y, tid = threadIdx.x; // 256
  const float hw = log1pf(expf(hwp[h])) * 0.13608276348795434f;
  const float rowq = -0.5f * hw * q2[i*NH + h];
  const float mi = mask[i];
  const long base = ((long)h*NRES + i) * NRES;
  const int j0 = tid, j1 = tid + 256;
  float v0 = logits[base+j0] + 0.57735026919f*biasb[((long)i*NRES+j0)*NH + h]
           + rowq - 0.5f*hw*k2[j0*NH+h] + INFV*(mi*mask[j0]-1.f);
  float v1 = logits[base+j1] + 0.57735026919f*biasb[((long)i*NRES+j1)*NH + h]
           + rowq - 0.5f*hw*k2[j1*NH+h] + INFV*(mi*mask[j1]-1.f);
  __shared__ float red[256];
  red[tid] = fmaxf(v0, v1); __syncthreads();
  for (int o=128;o>0;o>>=1){ if (tid<o) red[tid]=fmaxf(red[tid],red[tid+o]); __syncthreads(); }
  const float mx = red[0]; __syncthreads();
  float e0 = expf(v0-mx), e1 = expf(v1-mx);
  red[tid] = e0+e1; __syncthreads();
  for (int o=128;o>0;o>>=1){ if (tid<o) red[tid]+=red[tid+o]; __syncthreads(); }
  const float inv = 1.f / red[0];
  abf[base+j0] = (__bf16)(e0*inv);
  abf[base+j1] = (__bf16)(e1*inv);
}

// ---------------------------------------------------------------------------
// Assemble IPA concat features (bf16, 2112 wide) with inverse-frame points.
// ---------------------------------------------------------------------------
__global__ void build_feats(const float* __restrict__ O1,   // [h][512][48]
                            const float* __restrict__ O2,   // [i][12][128]
                            const float* __restrict__ Rbuf,
                            const float* __restrict__ tvec,
                            __bf16* __restrict__ feats)
{
  const int i = blockIdx.x;
  __shared__ float R[9], t3[3];
  if (threadIdx.x < 9) R[threadIdx.x] = Rbuf[i*9+threadIdx.x];
  if (threadIdx.x < 3) t3[threadIdx.x] = tvec[i*3+threadIdx.x];
  __syncthreads();
  for (int idx = threadIdx.x; idx < 2112; idx += 128){
    float v;
    if (idx < 192){
      int h = idx>>4, c = idx&15;
      v = O1[((long)h*NRES+i)*48 + c];
    } else if (idx < 576){
      int r = idx - 192;
      int seg = r / 96;           // 0,1,2 -> coords; 3 -> norm
      int rr = r % 96;
      int h = rr>>3, p = rr&7;
      const float* o = &O1[((long)h*NRES+i)*48 + 16 + p*3];
      float gx=o[0]-t3[0], gy=o[1]-t3[1], gz=o[2]-t3[2];
      float lx = R[0]*gx + R[3]*gy + R[6]*gz;   // R^T * g
      float ly = R[1]*gx + R[4]*gy + R[7]*gz;
      float lz = R[2]*gx + R[5]*gy + R[8]*gz;
      if (seg==0) v = lx; else if (seg==1) v = ly; else if (seg==2) v = lz;
      else v = sqrtf(lx*lx+ly*ly+lz*lz + EPSV);
    } else {
      int r = idx - 576;
      int h = r >> 7, c = r & 127;
      v = O2[((long)i*NH + h)*CZ + c];
    }
    feats[(long)i*2112 + idx] = (__bf16)v;
  }
}

// ---------------------------------------------------------------------------
// Backbone frame update: t += R*upd[3:6]; q = normalize(q + q*vec(upd[0:3]))
// ---------------------------------------------------------------------------
__global__ void update_frames(const float* __restrict__ upd,
                              float* __restrict__ quat, float* __restrict__ tvec,
                              const float* __restrict__ Rbuf,
                              float* __restrict__ Rb, float* __restrict__ tb,
                              float* __restrict__ out7)
{
  int i = blockIdx.x*128 + threadIdx.x;
  if (i >= NRES) return;
  const float* u = upd + i*6;
  const float* R = Rbuf + i*9;
  float tx = tvec[i*3+0] + R[0]*u[3]+R[1]*u[4]+R[2]*u[5];
  float ty = tvec[i*3+1] + R[3]*u[3]+R[4]*u[4]+R[5]*u[5];
  float tz = tvec[i*3+2] + R[6]*u[3]+R[7]*u[4]+R[8]*u[5];
  float w=quat[i*4], x=quat[i*4+1], y=quat[i*4+2], z=quat[i*4+3];
  float b=u[0], c=u[1], d=u[2];
  float nw = w - x*b - y*c - z*d;
  float nx = x + w*b + y*d - z*c;
  float ny = y + w*c - x*d + z*b;
  float nz = z + w*d + x*c - y*b;
  float inv = rsqrtf(nw*nw+nx*nx+ny*ny+nz*nz);
  nw*=inv; nx*=inv; ny*=inv; nz*=inv;
  quat[i*4]=nw; quat[i*4+1]=nx; quat[i*4+2]=ny; quat[i*4+3]=nz;
  tvec[i*3]=tx; tvec[i*3+1]=ty; tvec[i*3+2]=tz;
  float Rn[9];
  Rn[0]=1-2*(ny*ny+nz*nz); Rn[1]=2*(nx*ny-nw*nz); Rn[2]=2*(nx*nz+nw*ny);
  Rn[3]=2*(nx*ny+nw*nz);   Rn[4]=1-2*(nx*nx+nz*nz); Rn[5]=2*(ny*nz-nw*nx);
  Rn[6]=2*(nx*nz-nw*ny);   Rn[7]=2*(ny*nz+nw*nx);   Rn[8]=1-2*(nx*nx+ny*ny);
  #pragma unroll
  for (int r=0;r<9;++r) Rb[i*9+r]=Rn[r];
  tb[i*3]=tx*TSF; tb[i*3+1]=ty*TSF; tb[i*3+2]=tz*TSF;
  out7[i*7+0]=nw; out7[i*7+1]=nx; out7[i*7+2]=ny; out7[i*7+3]=nz;
  out7[i*7+4]=tx*TSF; out7[i*7+5]=ty*TSF; out7[i*7+6]=tz*TSF;
}

__global__ void ang_norm(const float* __restrict__ raw, float* __restrict__ o_un,
                         float* __restrict__ o_ang, float* __restrict__ angbuf)
{
  int idx = blockIdx.x*256 + threadIdx.x;
  if (idx >= NRES*7) return;
  int i = idx/7, a = idx%7;
  float x = raw[i*14 + a*2], y = raw[i*14 + a*2 + 1];
  o_un[idx*2]=x; o_un[idx*2+1]=y;
  float inv = rsqrtf(fmaxf(x*x+y*y, EPSV));
  float nx=x*inv, ny=y*inv;
  o_ang[idx*2]=nx; o_ang[idx*2+1]=ny;
  angbuf[idx*2]=nx; angbuf[idx*2+1]=ny;
}

// ---------------------------------------------------------------------------
// Torsion frames (chain-composed) + atom14 placement, one residue per thread.
// ---------------------------------------------------------------------------
__global__ void torsion_atoms(const float* __restrict__ Rb, const float* __restrict__ tb,
                              const float* __restrict__ angbuf, const int* __restrict__ aatype,
                              const float* __restrict__ defr, const int* __restrict__ gidx,
                              const float* __restrict__ amask, const float* __restrict__ litpos,
                              float* __restrict__ out_side, float* __restrict__ out_pos)
{
  int i = blockIdx.x*64 + threadIdx.x;
  if (i >= NRES) return;
  int aa = aatype[i];
  float fR[8][9], ft[8][3];
  for (int k=0;k<8;++k){
    const float* d4 = defr + ((long)aa*8 + k)*16;
    float s_, c_;
    if (k==0){ s_=0.f; c_=1.f; }
    else { s_=angbuf[(i*7 + (k-1))*2]; c_=angbuf[(i*7 + (k-1))*2 + 1]; }
    #pragma unroll
    for (int r=0;r<3;++r){
      float d0=d4[r*4+0], d1=d4[r*4+1], d2=d4[r*4+2];
      fR[k][r*3+0]=d0;
      fR[k][r*3+1]=d1*c_ + d2*s_;
      fR[k][r*3+2]=-d1*s_ + d2*c_;
      ft[k][r]=d4[r*4+3];
    }
  }
  auto mm = [](float* o, const float* a, const float* b){
    #pragma unroll
    for (int r=0;r<3;++r)
      #pragma unroll
      for (int c=0;c<3;++c)
        o[r*3+c]=a[r*3+0]*b[c]+a[r*3+1]*b[3+c]+a[r*3+2]*b[6+c];
  };
  auto mv = [](float* o, const float* a, const float* v){
    #pragma unroll
    for (int r=0;r<3;++r) o[r]=a[r*3]*v[0]+a[r*3+1]*v[1]+a[r*3+2]*v[2];
  };
  float tR[9], tt[3];
  for (int k=5;k<8;++k){
    mm(tR, fR[k-1], fR[k]); mv(tt, fR[k-1], ft[k]);
    #pragma unroll
    for (int r=0;r<9;++r) fR[k][r]=tR[r];
    #pragma unroll
    for (int r=0;r<3;++r) ft[k][r]=tt[r]+ft[k-1][r];
  }
  const float* Rbi = Rb + i*9; const float* tbi = tb + i*3;
  float Rg[8][9], tg[8][3];
  for (int k=0;k<8;++k){
    mm(Rg[k], Rbi, fR[k]); mv(tg[k], Rbi, ft[k]);
    #pragma unroll
    for (int r=0;r<3;++r) tg[k][r]+=tbi[r];
    float* o = out_side + ((long)i*8+k)*16;
    #pragma unroll
    for (int r=0;r<3;++r){
      o[r*4+0]=Rg[k][r*3+0]; o[r*4+1]=Rg[k][r*3+1]; o[r*4+2]=Rg[k][r*3+2]; o[r*4+3]=tg[k][r];
    }
    o[12]=0.f; o[13]=0.f; o[14]=0.f; o[15]=1.f;
  }
  for (int a=0;a<14;++a){
    int k = gidx[aa*14+a];
    const float* lp = litpos + ((long)aa*14+a)*3;
    float m = amask[aa*14+a];
    float px = Rg[k][0]*lp[0]+Rg[k][1]*lp[1]+Rg[k][2]*lp[2]+tg[k][0];
    float py = Rg[k][3]*lp[0]+Rg[k][4]*lp[1]+Rg[k][5]*lp[2]+tg[k][1];
    float pz = Rg[k][6]*lp[0]+Rg[k][7]*lp[1]+Rg[k][8]*lp[2]+tg[k][2];
    out_pos[((long)i*14+a)*3+0]=px*m;
    out_pos[((long)i*14+a)*3+1]=py*m;
    out_pos[((long)i*14+a)*3+2]=pz*m;
  }
}

// ===========================================================================
extern "C" void kernel_launch(void* const* d_in, const int* in_sizes, int n_in,
                              void* d_out, int out_size, void* d_ws, size_t ws_size,
                              hipStream_t stream)
{
  (void)in_sizes; (void)n_in; (void)out_size; (void)ws_size;
  const float* single = (const float*)d_in[0];
  const float* pair   = (const float*)d_in[1];
  const int*   aatype = (const int*)  d_in[2];
  const float* mask   = (const float*)d_in[3];
  const float* ln_s_g=(const float*)d_in[4],  *ln_s_b=(const float*)d_in[5];
  const float* ln_z_g=(const float*)d_in[6],  *ln_z_b=(const float*)d_in[7];
  const float* lin_w =(const float*)d_in[8],  *lin_b =(const float*)d_in[9];
  const float* q_w   =(const float*)d_in[10], *q_b   =(const float*)d_in[11];
  const float* kv_w  =(const float*)d_in[12], *kv_b  =(const float*)d_in[13];
  const float* qp_w  =(const float*)d_in[14], *qp_b  =(const float*)d_in[15];
  const float* kvp_w =(const float*)d_in[16], *kvp_b =(const float*)d_in[17];
  const float* b_w   =(const float*)d_in[18], *b_b   =(const float*)d_in[19];
  const float* hw_p  =(const float*)d_in[20];
  const float* out_w =(const float*)d_in[21], *out_b =(const float*)d_in[22];
  const float* lni_g =(const float*)d_in[23], *lni_b =(const float*)d_in[24];
  const float* tr_w1 =(const float*)d_in[25], *tr_b1 =(const float*)d_in[26];
  const float* tr_w2 =(const float*)d_in[27], *tr_b2 =(const float*)d_in[28];
  const float* tr_w3 =(const float*)d_in[29], *tr_b3 =(const float*)d_in[30];
  const float* lnt_g =(const float*)d_in[31], *lnt_b =(const float*)d_in[32];
  const float* bb_w  =(const float*)d_in[33], *bb_b  =(const float*)d_in[34];
  const float* ai_w  =(const float*)d_in[35], *ai_b  =(const float*)d_in[36];
  const float* an_w  =(const float*)d_in[37], *an_b  =(const float*)d_in[38];
  const float* ab1_w =(const float*)d_in[39], *ab1_b =(const float*)d_in[40];
  const float* ab2_w =(const float*)d_in[41], *ab2_b =(const float*)d_in[42];
  const float* ao_w  =(const float*)d_in[43], *ao_b  =(const float*)d_in[44];
  const float* defr  =(const float*)d_in[45];
  const int*   gidx  =(const int*)  d_in[46];
  const float* amaskp=(const float*)d_in[47];
  const float* litpos=(const float*)d_in[48];

  float* out = (float*)d_out;
  float* OUT_frames = out;                 // 8*512*7
  float* OUT_side   = out + 28672;         // 8*512*8*16
  float* OUT_unn    = out + 552960;        // 8*512*14
  float* OUT_ang    = out + 610304;        // 8*512*14
  float* OUT_pos    = out + 667648;        // 8*512*42
  float* OUT_sstk   = out + 839680;        // 8*512*384
  float* OUT_sfin   = out + 2412544;       // 512*384

  char* wsp = (char*)d_ws;
  size_t off = 0;
  auto alloc = [&](size_t bytes)->void*{
    void* p = (void*)(wsp + off);
    off += (bytes + 255) & ~(size_t)255;
    return p;
  };
  // f32 state
  float* s       = (float*)alloc((size_t)NRES*CS*4);
  float* s_norm  = (float*)alloc((size_t)NRES*CS*4);
  float* biasb   = (float*)alloc((size_t)NRES*NRES*NH*4);
  float* a_init  = (float*)alloc((size_t)NRES*CZ*4);
  float* qry     = (float*)alloc((size_t)NRES*192*4);
  float* kvf     = (float*)alloc((size_t)NRES*384*4);
  float* qpf     = (float*)alloc((size_t)NRES*144*4);
  float* kvpf    = (float*)alloc((size_t)NRES*432*4);
  float* quat    = (float*)alloc((size_t)NRES*4*4);
  float* tvec    = (float*)alloc((size_t)NRES*3*4);
  float* Rbuf    = (float*)alloc((size_t)NRES*9*4);
  float* Rbq     = (float*)alloc((size_t)NRES*9*4);
  float* tbv     = (float*)alloc((size_t)NRES*3*4);
  float* q2      = (float*)alloc((size_t)NRES*NH*4);
  float* k2      = (float*)alloc((size_t)NRES*NH*4);
  float* logits  = (float*)alloc((size_t)NH*NRES*NRES*4);
  float* O1      = (float*)alloc((size_t)NH*NRES*48*4);
  float* O2      = (float*)alloc((size_t)NRES*NH*CZ*4);
  float* tf1     = (float*)alloc((size_t)NRES*CS*4);
  float* tf2     = (float*)alloc((size_t)NRES*CS*4);
  float* updb    = (float*)alloc((size_t)NRES*6*4);
  float* hang    = (float*)alloc((size_t)NRES*CZ*4);
  float* rang    = (float*)alloc((size_t)NRES*CZ*4);
  float* angraw  = (float*)alloc((size_t)NRES*14*4);
  float* angbuf  = (float*)alloc((size_t)NRES*14*4);
  // bf16 activations
  __bf16* s_bf     = (__bf16*)alloc((size_t)NRES*CS*2);
  __bf16* snorm_bf = (__bf16*)alloc((size_t)NRES*CS*2);
  __bf16* sinit_bf = (__bf16*)alloc((size_t)NRES*CS*2);
  __bf16* z_bf     = (__bf16*)alloc((size_t)NRES*NRES*CZ*2);
  __bf16* z_T      = (__bf16*)alloc((size_t)NRES*CZ*NRES*2);
  __bf16* Qext     = (__bf16*)alloc((size_t)NH*NRES*32*2);
  __bf16* Kext     = (__bf16*)alloc((size_t)NH*NRES*32*2);
  __bf16* VVt      = (__bf16*)alloc((size_t)NH*48*NRES*2);
  __bf16* abf      = (__bf16*)alloc((size_t)16*NRES*NRES*2); // 16 head rows (4 pad)
  __bf16* feats    = (__bf16*)alloc((size_t)NRES*2112*2);
  __bf16* tb1      = (__bf16*)alloc((size_t)NRES*CS*2);
  __bf16* tb2      = (__bf16*)alloc((size_t)NRES*CS*2);
  // bf16 transposed weights
  __bf16* lin_t =(__bf16*)alloc((size_t)384*384*2);
  __bf16* qw_t  =(__bf16*)alloc((size_t)192*384*2);
  __bf16* kvw_t =(__bf16*)alloc((size_t)384*384*2);
  __bf16* qpw_t =(__bf16*)alloc((size_t)144*384*2);
  __bf16* kvpw_t=(__bf16*)alloc((size_t)432*384*2);
  __bf16* bw_t  =(__bf16*)alloc((size_t)16*128*2);
  __bf16* outw_t=(__bf16*)alloc((size_t)384*2112*2);
  __bf16* tr1_t =(__bf16*)alloc((size_t)384*384*2);
  __bf16* tr2_t =(__bf16*)alloc((size_t)384*384*2);
  __bf16* tr3_t =(__bf16*)alloc((size_t)384*384*2);
  __bf16* bbw_t =(__bf16*)alloc((size_t)16*384*2);
  __bf16* ait   =(__bf16*)alloc((size_t)128*384*2);
  __bf16* ant   =(__bf16*)alloc((size_t)128*384*2);
  __bf16* ab1t  =(__bf16*)alloc((size_t)2*128*128*2);
  __bf16* ab2t  =(__bf16*)alloc((size_t)2*128*128*2);
  __bf16* aot   =(__bf16*)alloc((size_t)16*128*2);

  auto gemm = [&](const __bf16* A, int lda, const __bf16* Bt, float* C, int ldc,
                  int M, int N, int Npad, int K, int batch,
                  long sA, long sB, long sC,
                  const float* bias, const float* res, int ldres, int relu){
    dim3 grid(cdiv_h(Npad,32), (M+15)/16, batch);
    gemm_bf16_wmma<<<grid, 32, 0, stream>>>(A,lda,Bt,C,ldc,M,N,K,Npad,sA,sB,sC,bias,res,ldres,relu);
  };
  auto conv = [&](const float* src, __bf16* dst, int n, int relu){
    conv_bf16_kernel<<<cdiv_h(n,256),256,0,stream>>>(src,dst,n,relu);
  };
  auto wt = [&](const float* src, __bf16* dst, int K, int N, int Np){
    wtrans_kernel<<<cdiv_h((long)Np*K,256),256,0,stream>>>(src,dst,K,N,Np);
  };

  // ---------------- pre-pass ----------------
  ln_kernel<<<NRES,128,0,stream>>>(single, ln_s_g, ln_s_b, s_norm, snorm_bf, CS);
  ln_kernel<<<NRES*NRES,128,0,stream>>>(pair, ln_z_g, ln_z_b, (float*)0, z_bf, CZ);
  conv(s_norm, sinit_bf, NRES*CS, 1);
  ztrans_kernel<<<cdiv_h((long)NRES*CZ*NRES,256),256,0,stream>>>(z_bf, z_T);
  // zero the 4 pad head-planes of abf so the o_pair GEMM can load rows 12..15
  zero_bf16_kernel<<<cdiv_h((long)4*NRES*NRES,256),256,0,stream>>>(
      abf + (size_t)NH*NRES*NRES, 4*NRES*NRES);
  wt(lin_w, lin_t, 384,384,384);
  wt(q_w, qw_t, 384,192,192);
  wt(kv_w, kvw_t, 384,384,384);
  wt(qp_w, qpw_t, 384,144,144);
  wt(kvp_w, kvpw_t, 384,432,432);
  wt(b_w, bw_t, 128,12,16);
  wt(out_w, outw_t, 2112,384,384);
  wt(tr_w1, tr1_t, 384,384,384);
  wt(tr_w2, tr2_t, 384,384,384);
  wt(tr_w3, tr3_t, 384,384,384);
  wt(bb_w, bbw_t, 384,6,16);
  wt(ai_w, ait, 384,128,128);
  wt(an_w, ant, 384,128,128);
  wt(ab1_w,          ab1t,          128,128,128);
  wt(ab1_w + 16384,  ab1t + 16384,  128,128,128);
  wt(ab2_w,          ab2t,          128,128,128);
  wt(ab2_w + 16384,  ab2t + 16384,  128,128,128);
  wt(ao_w, aot, 128,14,16);
  // pair bias (computed once; z is loop-invariant)
  gemm(z_bf,128, bw_t, biasb, 12, NRES*NRES,12,16,128, 1,0,0,0, b_b, 0,0,0);
  // angle-resnet initial projection (loop-invariant)
  gemm(sinit_bf,384, ait, a_init, 128, NRES,128,128,384, 1,0,0,0, ai_b, 0,0,0);
  // s = LN(single) @ lin_in + b
  gemm(snorm_bf,384, lin_t, s, 384, NRES,384,384,384, 1,0,0,0, lin_b, 0,0,0);
  conv(s, s_bf, NRES*CS, 0);
  init_qt<<<2,256,0,stream>>>(quat, tvec);

  // ---------------- 8 structure blocks ----------------
  for (int blk = 0; blk < 8; ++blk){
    gemm(s_bf,384, qw_t,  qry, 192, NRES,192,192,384, 1,0,0,0, q_b,  0,0,0);
    gemm(s_bf,384, kvw_t, kvf, 384, NRES,384,384,384, 1,0,0,0, kv_b, 0,0,0);
    gemm(s_bf,384, qpw_t, qpf, 144, NRES,144,144,384, 1,0,0,0, qp_b, 0,0,0);
    gemm(s_bf,384, kvpw_t,kvpf,432, NRES,432,432,384, 1,0,0,0, kvp_b,0,0,0);
    build_attn<<<NRES,64,0,stream>>>(qry,kvf,qpf,kvpf,quat,tvec,hw_p,Rbuf,Qext,Kext,q2,k2,VVt);
    // logits[h] = Qext_h @ Kext_h^T   (K=32 folds scalar-qk + hw-scaled point-qk)
    gemm(Qext,32, Kext, logits, NRES, NRES,NRES,NRES,32, NH,
         (long)NRES*32,(long)NRES*32,(long)NRES*NRES, 0,0,0,0);
    attn_softmax<<<dim3(NRES,NH),256,0,stream>>>(logits,biasb,q2,k2,hw_p,mask,abf);
    // [o | o_pt] per head
    gemm(abf,NRES, VVt, O1, 48, NRES,40,48,NRES, NH,
         (long)NRES*NRES,(long)48*NRES,(long)NRES*48, 0,0,0,0);
    // o_pair, batched per residue i: (12x512) @ z[i]^T(128x512); A rows 12..15 zero-padded
    gemm(abf, NRES*NRES, z_T, O2, CZ, NH,CZ,CZ,NRES, NRES,
         (long)NRES, (long)CZ*NRES, (long)NH*CZ, 0,0,0,0);
    build_feats<<<NRES,128,0,stream>>>(O1,O2,Rbuf,tvec,feats);
    gemm(feats,2112, outw_t, s, 384, NRES,384,384,2112, 1,0,0,0, out_b, s,384, 0);
    ln_kernel<<<NRES,128,0,stream>>>(s, lni_g, lni_b, s, s_bf, CS);
    // transition MLP + residual
    gemm(s_bf,384, tr1_t, tf1, 384, NRES,384,384,384, 1,0,0,0, tr_b1, 0,0, 1);
    conv(tf1, tb1, NRES*CS, 0);
    gemm(tb1,384, tr2_t, tf2, 384, NRES,384,384,384, 1,0,0,0, tr_b2, 0,0, 1);
    conv(tf2, tb2, NRES*CS, 0);
    gemm(tb2,384, tr3_t, s, 384, NRES,384,384,384, 1,0,0,0, tr_b3, s,384, 0);
    ln_kernel<<<NRES,128,0,stream>>>(s, lnt_g, lnt_b, s, s_bf, CS);
    // backbone update
    gemm(s_bf,384, bbw_t, updb, 6, NRES,6,16,384, 1,0,0,0, bb_b, 0,0,0);
    update_frames<<<4,128,0,stream>>>(updb,quat,tvec,Rbuf,Rbq,tbv,
                                      OUT_frames + (long)blk*NRES*7);
    // angle resnet
    conv(s, tb1, NRES*CS, 1);
    gemm(tb1,384, ant, hang, 128, NRES,128,128,384, 1,0,0,0, an_b, a_init,128, 0);
    for (int rb = 0; rb < 2; ++rb){
      conv(hang, tb2, NRES*CZ, 1);
      gemm(tb2,128, ab1t + rb*16384, rang, 128, NRES,128,128,128, 1,0,0,0, ab1_b + rb*128, 0,0,0);
      conv(rang, tb2, NRES*CZ, 1);
      gemm(tb2,128, ab2t + rb*16384, hang, 128, NRES,128,128,128, 1,0,0,0, ab2_b + rb*128, hang,128, 0);
    }
    conv(hang, tb2, NRES*CZ, 1);
    gemm(tb2,128, aot, angraw, 14, NRES,14,16,128, 1,0,0,0, ao_b, 0,0,0);
    ang_norm<<<cdiv_h(NRES*7,256),256,0,stream>>>(angraw,
        OUT_unn + (long)blk*NRES*14, OUT_ang + (long)blk*NRES*14, angbuf);
    torsion_atoms<<<8,64,0,stream>>>(Rbq,tbv,angbuf,aatype,defr,gidx,amaskp,litpos,
        OUT_side + (long)blk*NRES*128, OUT_pos + (long)blk*NRES*42);
    copy_kernel<<<cdiv_h(NRES*CS,256),256,0,stream>>>(s, OUT_sstk + (long)blk*NRES*CS, NRES*CS);
  }
  copy_kernel<<<cdiv_h(NRES*CS,256),256,0,stream>>>(s, OUT_sfin, NRES*CS);
}
